// Transformer_57363583205586
// MI455X (gfx1250) — matrix operand order repaired and checked by reading, not compiled, not run on hardware
//
#include <hip/hip_runtime.h>
#include <hip/hip_bf16.h>
#include <math.h>

// Problem constants (from reference)
#define NT 768      // tokens
#define CD 768      // channels
#define CSD 384     // single_cond channels
#define CPD 128     // pair channels
#define NH 16       // heads
#define HD 48       // head dim
#define FFD 1536    // FFN intermediate F
#define SBH 64      // S*H = 4*16

typedef __attribute__((ext_vector_type(16))) _Float16 v16h;
typedef __attribute__((ext_vector_type(8)))  _Float16 v8h;
typedef __attribute__((ext_vector_type(8)))  float    v8f;

__device__ __forceinline__ v8f wmma_f16(v16h a, v16h b, v8f c) {
  return __builtin_amdgcn_wmma_f32_16x16x32_f16(
      false, a, false, b, (short)0, c, false, false);
}
__device__ __forceinline__ float sigm(float x) { return 1.f / (1.f + expf(-x)); }
__device__ __forceinline__ v8h zero8() {
  v8h z;
  #pragma unroll
  for (int i = 0; i < 8; ++i) z[i] = (_Float16)0.f;
  return z;
}
// A fragment: two 16B chunks at p and p+16 halves (ISA 16-bit A layout)
__device__ __forceinline__ v16h ld_frag_a(const _Float16* p) {
  union { v16h v; v8h h[2]; } u;
  u.h[0] = *(const v8h*)(p);
  u.h[1] = *(const v8h*)(p + 16);
  return u.v;
}
// B fragment: two 16B chunks at p and p+16 halves (same k order as A)
__device__ __forceinline__ v16h ld_frag_b(const _Float16* p) {
  union { v16h v; v8h h[2]; } u;
  u.h[0] = *(const v8h*)(p);
  u.h[1] = *(const v8h*)(p + 16);
  return u.v;
}

// ---------------------------------------------------------------------------
// Weight pre-pass: BT[Nn][K] (f16) = transpose(B[K][Nn] f32), tiled via LDS.
// One pass per GEMM; removes per-M-tile transpose work from the hot loop.
// ---------------------------------------------------------------------------
__global__ __launch_bounds__(256) void k_wtr(
    const float* __restrict__ B, int ldb, int K,
    _Float16* __restrict__ BT)
{
  __shared__ float t[64][65];
  const int k0 = blockIdx.x * 64, n0 = blockIdx.y * 64;
  const int tid = threadIdx.x;
  #pragma unroll
  for (int i = 0; i < 16; ++i) {               // coalesced over n
    int idx = tid + i * 256; int r = idx >> 6, c = idx & 63;
    t[r][c] = B[(size_t)(k0 + r) * ldb + n0 + c];
  }
  __syncthreads();
  #pragma unroll
  for (int i = 0; i < 16; ++i) {               // coalesced over k
    int idx = tid + i * 256; int r = idx >> 6, c = idx & 63;
    BT[(size_t)(n0 + r) * K + k0 + c] = (_Float16)t[c][r];
  }
}

// ---------------------------------------------------------------------------
// Generic f16 WMMA GEMM: C[M,Nn] = A(f16)[M,K] x BT(f16)[Nn][K] (+bias)
// Block tile 64x128, 8 waves (2x4), wave tile 32x32, BK=32.
// EPI: 0=f32  1=f32 sigmoid  2=f16  3=Cf+=ep0*acc  4=f16 transposed (vT[c][n])
// ---------------------------------------------------------------------------
template<int EPI>
__global__ __launch_bounds__(256) void k_gemm(
    const _Float16* __restrict__ A, int lda,
    const _Float16* __restrict__ BT,
    const float* __restrict__ bias,
    float* __restrict__ Cf, _Float16* __restrict__ Ch, int ldc,
    const float* __restrict__ ep0, int K)
{
  __shared__ alignas(16) _Float16 As[64][32];
  __shared__ alignas(16) _Float16 Bs[128][40];   // fragment-major: Bs[n][k]
  const int tid = threadIdx.x;
  const int lane = tid & 31, wid = tid >> 5;
  const int wm = wid >> 2, wn = wid & 3;         // 2x4 wave grid
  const int m0 = blockIdx.y * 64, n0 = blockIdx.x * 128;
  const int lr = lane & 15, hs = lane >> 4;
  v8f z = {0.f,0.f,0.f,0.f,0.f,0.f,0.f,0.f};
  v8f acc[2][2] = {{z,z},{z,z}};

  for (int k0 = 0; k0 < K; k0 += 32) {
    { // A tile: one 128-bit copy per thread
      int base = tid * 8; int r = base >> 5, c = base & 31;
      *(v8h*)&As[r][c] = *(const v8h*)(A + (size_t)(m0 + r) * lda + k0 + c);
    }
    { // B tile: two 128-bit copies per thread (already f16, transposed)
      int r = tid >> 1, hsl = (tid & 1) * 16;
      const _Float16* src = BT + (size_t)(n0 + r) * K + k0 + hsl;
      *(v8h*)&Bs[r][hsl]     = *(const v8h*)(src);
      *(v8h*)&Bs[r][hsl + 8] = *(const v8h*)(src + 8);
    }
    if (k0 + 32 < K)  // gfx1250 global_prefetch_b8 hint
      __builtin_prefetch(BT + (size_t)(n0 + (tid >> 1)) * K + k0 + 32, 0, 1);
    __syncthreads();

    v16h af[2];
    af[0] = ld_frag_a(&As[wm * 32 + lr][hs * 8]);
    af[1] = ld_frag_a(&As[wm * 32 + 16 + lr][hs * 8]);
    #pragma unroll
    for (int ni = 0; ni < 2; ++ni) {
      v16h bf = ld_frag_b(&Bs[wn * 32 + ni * 16 + lr][hs * 8]);
      acc[0][ni] = wmma_f16(af[0], bf, acc[0][ni]);
      acc[1][ni] = wmma_f16(af[1], bf, acc[1][ni]);
    }
    __syncthreads();
  }

  #pragma unroll
  for (int mi = 0; mi < 2; ++mi)
    #pragma unroll
    for (int ni = 0; ni < 2; ++ni)
      #pragma unroll
      for (int r = 0; r < 8; ++r) {
        int row = m0 + wm * 32 + mi * 16 + hs * 8 + r;
        int col = n0 + wn * 32 + ni * 16 + lr;
        float v = acc[mi][ni][r];
        if (bias) v += bias[col];
        size_t idx = (size_t)row * ldc + col;
        if (EPI == 0)      Cf[idx] = v;
        else if (EPI == 1) Cf[idx] = sigm(v);
        else if (EPI == 2) Ch[idx] = (_Float16)v;
        else if (EPI == 3) Cf[idx] += ep0[idx] * v;
        else               Ch[(size_t)col * NT + row] = (_Float16)v;  // vT
      }
}

// ---------------------------------------------------------------------------
// Pair-logit GEMM with fused row LayerNorm, transposed output:
//   plT[s*16+h][qk] = LN(pair_cond[qk,:]) * pair_ln_scale  @  pwT
// ---------------------------------------------------------------------------
__global__ __launch_bounds__(256) void k_pl(
    const float* __restrict__ pair_cond,   // [N*N, 128]
    const float* __restrict__ pls,         // [128]
    const _Float16* __restrict__ pwT,      // [64][128] pre-transposed f16
    _Float16* __restrict__ plT)            // [64, N*N]
{
  __shared__ alignas(16) _Float16 Ah[64][128];
  __shared__ alignas(16) _Float16 Bh[64][136];   // fragment-major: Bh[sh][k]
  __shared__ float ps[64][4], ps2[64][4], mu[64], rs[64];
  const int tid = threadIdx.x;
  const size_t qk0 = (size_t)blockIdx.x * 64;

  // per-row stats; keep the 32 inputs in registers for the normalize pass
  const int ar = tid >> 2, aq = tid & 3;
  float fv[32];
  {
    const float* src = pair_cond + (qk0 + ar) * 128 + aq * 32;
    float s = 0.f, s2 = 0.f;
    #pragma unroll
    for (int i = 0; i < 32; ++i) { float v = src[i]; fv[i] = v; s += v; s2 += v * v; }
    ps[ar][aq] = s; ps2[ar][aq] = s2;
  }
  { // B tile: four 128-bit copies per thread
    int c = tid >> 2, sl = (tid & 3) * 32;
    const _Float16* src = pwT + (size_t)c * 128 + sl;
    #pragma unroll
    for (int j = 0; j < 4; ++j)
      *(v8h*)&Bh[c][sl + j * 8] = *(const v8h*)(src + j * 8);
  }
  __syncthreads();
  if (tid < 64) {
    float s  = ps[tid][0] + ps[tid][1] + ps[tid][2] + ps[tid][3];
    float s2 = ps2[tid][0] + ps2[tid][1] + ps2[tid][2] + ps2[tid][3];
    float m = s * (1.f / 128.f);
    float var = s2 * (1.f / 128.f) - m * m;
    mu[tid] = m; rs[tid] = rsqrtf(var + 1e-5f);
  }
  __syncthreads();
  { // normalize -> f16 A tile (contiguous 32-half runs per thread)
    float m = mu[ar], r = rs[ar];
    #pragma unroll
    for (int i = 0; i < 32; ++i)
      Ah[ar][aq * 32 + i] = (_Float16)((fv[i] - m) * r * pls[aq * 32 + i]);
  }
  __syncthreads();

  const int lane = tid & 31, wid = tid >> 5;
  const int wr = (wid >> 2) * 32, wc = (wid & 3) * 16;
  const int lr = lane & 15, hs = lane >> 4;
  v8f z = {0.f,0.f,0.f,0.f,0.f,0.f,0.f,0.f};
  v8f acc[2] = {z, z};
  for (int kk = 0; kk < 128; kk += 32) {
    v16h bf = ld_frag_b(&Bh[wc + lr][kk + hs * 8]);
    #pragma unroll
    for (int mi = 0; mi < 2; ++mi) {
      v16h a = ld_frag_a(&Ah[wr + mi * 16 + lr][kk + hs * 8]);
      acc[mi] = wmma_f16(a, bf, acc[mi]);
    }
  }
  #pragma unroll
  for (int mi = 0; mi < 2; ++mi) {
    int col = wc + lr;
    _Float16* dst = plT + (size_t)col * (NT * NT) + qk0 + wr + mi * 16 + hs * 8;
    #pragma unroll
    for (int r = 0; r < 8; ++r) dst[r] = (_Float16)acc[mi][r]; // contiguous
  }
}

// ---------------------------------------------------------------------------
// Attention scores: logits[h,q,k] = (q.k)/sqrt(48) + 1e9*(mask[k]-1) + plT
// ---------------------------------------------------------------------------
__global__ __launch_bounds__(256) void k_scores(
    const _Float16* __restrict__ q16, const _Float16* __restrict__ k16,
    const float* __restrict__ mask, const _Float16* __restrict__ plT,
    int s, float* __restrict__ logits)
{
  __shared__ alignas(16) _Float16 Aq[64][64];
  __shared__ alignas(16) _Float16 Bk[64][72];   // Bk[kcol][d]
  const int h = blockIdx.z;
  const int q0 = blockIdx.y * 64, k0 = blockIdx.x * 64;
  const int tid = threadIdx.x;
  { // one v8h per (row, slot); d=0..23 loaded, d=24..31 zeroed
    int r = tid >> 2, sl = tid & 3;
    v8h va = zero8(), vb = zero8();
    if (sl < 3) {
      va = *(const v8h*)(q16 + (size_t)(q0 + r) * CD + h * HD + sl * 8);
      vb = *(const v8h*)(k16 + (size_t)(k0 + r) * CD + h * HD + sl * 8);
    }
    *(v8h*)&Aq[r][sl * 8] = va;
    *(v8h*)&Bk[r][sl * 8] = vb;
  }
  __syncthreads();
  const int lane = tid & 31, wid = tid >> 5;
  const int wr = (wid >> 2) * 32, wc = (wid & 3) * 16;
  const int lr = lane & 15, hs = lane >> 4;
  v8f z = {0.f,0.f,0.f,0.f,0.f,0.f,0.f,0.f};
  v8f acc[2] = {z, z};
  #pragma unroll
  for (int kk = 0; kk < 64; kk += 32) {
    v16h bf = ld_frag_b(&Bk[wc + lr][kk + hs * 8]);
    #pragma unroll
    for (int mi = 0; mi < 2; ++mi) {
      v16h a = ld_frag_a(&Aq[wr + mi * 16 + lr][kk + hs * 8]);
      acc[mi] = wmma_f16(a, bf, acc[mi]);
    }
  }
  const float scale = 0.14433756729740643f; // 1/sqrt(48)
  const _Float16* pl = plT + (size_t)(s * 16 + h) * (NT * NT);
  #pragma unroll
  for (int mi = 0; mi < 2; ++mi)
    #pragma unroll
    for (int r = 0; r < 8; ++r) {
      int qq = q0 + wr + mi * 16 + hs * 8 + r;
      int kk = k0 + wc + lr;
      float b = 1e9f * (mask[kk] - 1.0f);
      float pv = (float)pl[(size_t)qq * NT + kk];
      logits[((size_t)h * NT + qq) * NT + kk] = acc[mi][r] * scale + b + pv;
    }
}

// Row softmax over k (768); f16 output
__global__ __launch_bounds__(256) void k_softmax(
    const float* __restrict__ logits, _Float16* __restrict__ w)
{
  const size_t row = blockIdx.x;
  const float* x = logits + row * NT;
  __shared__ float red[256];
  const int tid = threadIdx.x;
  float v0 = x[tid], v1 = x[tid + 256], v2 = x[tid + 512];
  red[tid] = fmaxf(v0, fmaxf(v1, v2)); __syncthreads();
  for (int t = 128; t > 0; t >>= 1) { if (tid < t) red[tid] = fmaxf(red[tid], red[tid + t]); __syncthreads(); }
  float m = red[0]; __syncthreads();
  float e0 = expf(v0 - m), e1 = expf(v1 - m), e2 = expf(v2 - m);
  red[tid] = e0 + e1 + e2; __syncthreads();
  for (int t = 128; t > 0; t >>= 1) { if (tid < t) red[tid] += red[tid + t]; __syncthreads(); }
  float inv = 1.f / red[0];
  w[row * NT + tid]       = (_Float16)(e0 * inv);
  w[row * NT + tid + 256] = (_Float16)(e1 * inv);
  w[row * NT + tid + 512] = (_Float16)(e2 * inv);
}

// Per-head weighted average with fused gating; V pre-transposed (vT[c][n]):
//   wa16[q, h*48+d] = f16( gate[q,h*48+d] * sum_k wts[h,q,k] * vT[h*48+d][k] )
__global__ __launch_bounds__(256) void k_wa(
    const _Float16* __restrict__ wts16, const _Float16* __restrict__ vT,
    const float* __restrict__ gate, _Float16* __restrict__ wa16)
{
  __shared__ alignas(16) _Float16 Aw[64][32];
  __shared__ alignas(16) _Float16 Bv[64][40];   // Bv[d][k]
  const int h = blockIdx.y;
  const int m0 = blockIdx.x * 64;
  const int tid = threadIdx.x;
  const int lane = tid & 31, wid = tid >> 5;
  const int wr = (wid >> 2) * 32, wc = (wid & 3) * 16;
  const int lr = lane & 15, hs = lane >> 4;
  v8f z = {0.f,0.f,0.f,0.f,0.f,0.f,0.f,0.f};
  v8f acc[2] = {z, z};
  for (int k0 = 0; k0 < NT; k0 += 32) {
    { int base = tid * 8; int r = base >> 5, c = base & 31;
      *(v8h*)&Aw[r][c] =
          *(const v8h*)(wts16 + ((size_t)h * NT + m0 + r) * NT + k0 + c); }
    { int c = tid >> 2, sl = tid & 3;
      v8h val = zero8();
      if (c < HD) val = *(const v8h*)(vT + (size_t)(h * HD + c) * NT + k0 + sl * 8);
      *(v8h*)&Bv[c][sl * 8] = val; }
    __syncthreads();
    v16h bf = ld_frag_b(&Bv[wc + lr][hs * 8]);
    #pragma unroll
    for (int mi = 0; mi < 2; ++mi) {
      v16h a = ld_frag_a(&Aw[wr + mi * 16 + lr][hs * 8]);
      acc[mi] = wmma_f16(a, bf, acc[mi]);
    }
    __syncthreads();
  }
  #pragma unroll
  for (int mi = 0; mi < 2; ++mi)
    #pragma unroll
    for (int r = 0; r < 8; ++r) {
      int col = wc + lr;
      if (col < HD) {
        int row = m0 + wr + mi * 16 + hs * 8 + r;
        size_t idx = (size_t)row * CD + h * HD + col;
        wa16[idx] = (_Float16)(acc[mi][r] * gate[idx]);
      }
    }
}

// LayerNorm rows of X[rows, width]; optional scale vec; f32 and/or f16 out
__global__ __launch_bounds__(256) void k_ln(
    const float* __restrict__ X, int width, const float* __restrict__ scale,
    float* __restrict__ outF, _Float16* __restrict__ outH)
{
  const size_t row = blockIdx.x;
  const float* x = X + row * width;
  __shared__ float red[256];
  const int tid = threadIdx.x;
  float s = 0.f;
  for (int i = tid; i < width; i += 256) s += x[i];
  red[tid] = s; __syncthreads();
  for (int t = 128; t > 0; t >>= 1) { if (tid < t) red[tid] += red[tid + t]; __syncthreads(); }
  float m = red[0] / width; __syncthreads();
  float s2 = 0.f;
  for (int i = tid; i < width; i += 256) { float d = x[i] - m; s2 += d * d; }
  red[tid] = s2; __syncthreads();
  for (int t = 128; t > 0; t >>= 1) { if (tid < t) red[tid] += red[tid + t]; __syncthreads(); }
  float inv = rsqrtf(red[0] / width + 1e-5f);
  for (int i = tid; i < width; i += 256) {
    float y = (x[i] - m) * inv;
    if (scale) y *= scale[i];
    if (outF) outF[row * width + i] = y;
    if (outH) outH[row * width + i] = (_Float16)y;
  }
}

__global__ void k_cvt16(_Float16* __restrict__ o, const float* __restrict__ x) {
  size_t i = (size_t)blockIdx.x * 256 + threadIdx.x;
  o[i] = (_Float16)x[i];
}
__global__ void k_make_x(_Float16* __restrict__ x16, const float* __restrict__ g1,
                         const float* __restrict__ lnA, const float* __restrict__ b1) {
  size_t i = (size_t)blockIdx.x * 256 + threadIdx.x;
  x16[i] = (_Float16)(g1[i] * lnA[i] + b1[i]);
}
__global__ void k_glu(_Float16* __restrict__ c16, const float* __restrict__ u) {
  size_t i = (size_t)blockIdx.x * 256 + threadIdx.x;
  size_t n = i / FFD, f = i % FFD;
  float u1 = u[n * (2 * FFD) + f], u2 = u[n * (2 * FFD) + FFD + f];
  c16[i] = (_Float16)(u1 * sigm(u1) * u2);
}

// ---------------------------------------------------------------------------
struct GemmCtx { _Float16* bt; hipStream_t st; };

static void gemm(GemmCtx g, int epi, const _Float16* A, int lda,
                 const float* B, int ldb, const float* bias,
                 float* Cf, _Float16* Ch, int ldc,
                 const float* ep0, int M, int Nn, int K)
{
  // pre-transpose weights to f16 BT[Nn][K] (single pass, L2-resident)
  k_wtr<<<dim3(K / 64, Nn / 64), 256, 0, g.st>>>(B, ldb, K, g.bt);
  dim3 grid(Nn / 128, M / 64);
  if (epi == 0)      k_gemm<0><<<grid, 256, 0, g.st>>>(A, lda, g.bt, bias, Cf, Ch, ldc, ep0, K);
  else if (epi == 1) k_gemm<1><<<grid, 256, 0, g.st>>>(A, lda, g.bt, bias, Cf, Ch, ldc, ep0, K);
  else if (epi == 2) k_gemm<2><<<grid, 256, 0, g.st>>>(A, lda, g.bt, bias, Cf, Ch, ldc, ep0, K);
  else if (epi == 3) k_gemm<3><<<grid, 256, 0, g.st>>>(A, lda, g.bt, bias, Cf, Ch, ldc, ep0, K);
  else               k_gemm<4><<<grid, 256, 0, g.st>>>(A, lda, g.bt, bias, Cf, Ch, ldc, ep0, K);
}

extern "C" void kernel_launch(void* const* d_in, const int* in_sizes, int n_in,
                              void* d_out, int out_size, void* d_ws, size_t ws_size,
                              hipStream_t stream)
{
  (void)in_sizes; (void)n_in; (void)out_size; (void)ws_size;
  const float* act         = (const float*)d_in[0];
  const float* mask        = (const float*)d_in[1];
  const float* single_cond = (const float*)d_in[2];
  const float* pair_cond   = (const float*)d_in[3];
  const float* attn_cln    = (const float*)d_in[4];
  const float* attn_cs_w   = (const float*)d_in[5];
  const float* attn_cs_b   = (const float*)d_in[6];
  const float* attn_cb_w   = (const float*)d_in[7];
  const float* q_w  = (const float*)d_in[8];
  const float* q_b  = (const float*)d_in[9];
  const float* k_w  = (const float*)d_in[10];
  const float* v_w  = (const float*)d_in[11];
  const float* g_w  = (const float*)d_in[12];
  const float* g_b  = (const float*)d_in[13];
  const float* o_w  = (const float*)d_in[14];
  const float* az_w = (const float*)d_in[15];
  const float* az_b = (const float*)d_in[16];
  const float* f_cln  = (const float*)d_in[17];
  const float* f_cs_w = (const float*)d_in[18];
  const float* f_cs_b = (const float*)d_in[19];
  const float* f_cb_w = (const float*)d_in[20];
  const float* t1_w   = (const float*)d_in[21];
  const float* t2_w   = (const float*)d_in[22];
  const float* f_az_w = (const float*)d_in[23];
  const float* f_az_b = (const float*)d_in[24];
  const float* pair_ln = (const float*)d_in[25];
  const float* pair_w  = (const float*)d_in[26];

  float* a = (float*)d_out;   // residual stream lives in d_out

  char* ws = (char*)d_ws;
  size_t off = 0;
  auto bump = [&](size_t bytes) -> char* {
    char* p = ws + off; off = (off + bytes + 255) & ~(size_t)255; return p;
  };
  _Float16* sc16  = (_Float16*)bump((size_t)NT * CSD * 2);
  _Float16* cn16  = (_Float16*)bump((size_t)NT * CSD * 2);
  float*    lnA   = (float*)bump((size_t)NT * CD * 4);
  float*    s1    = (float*)bump((size_t)NT * CD * 4);
  float*    b1    = (float*)bump((size_t)NT * CD * 4);
  _Float16* x16   = (_Float16*)bump((size_t)NT * CD * 2);
  _Float16* q16   = (_Float16*)bump((size_t)NT * CD * 2);
  _Float16* k16   = (_Float16*)bump((size_t)NT * CD * 2);
  _Float16* vT16  = (_Float16*)bump((size_t)NT * CD * 2);   // transposed V
  float*    gate  = (float*)bump((size_t)NT * CD * 4);
  float*    az    = (float*)bump((size_t)NT * CD * 4);
  _Float16* bt16  = (_Float16*)bump((size_t)CD * 2 * FFD * 2); // BT scratch (max 768x3072)
  _Float16* pwT16 = (_Float16*)bump((size_t)SBH * CPD * 2);    // pwT [64][128]
  float*    logit = (float*)bump((size_t)NH * NT * NT * 4);
  _Float16* wts16 = (_Float16*)bump((size_t)NH * NT * NT * 2);
  _Float16* wa16  = (_Float16*)bump((size_t)NT * CD * 2);
  float*    u     = (float*)bump((size_t)NT * 2 * FFD * 4);
  _Float16* c16   = (_Float16*)bump((size_t)NT * FFD * 2);
  _Float16* plT16 = (_Float16*)bump((size_t)SBH * NT * NT * 2); // [s*16+h][q*N+k]

  GemmCtx G{bt16, stream};

  hipMemcpyAsync(a, act, (size_t)NT * CD * 4, hipMemcpyDeviceToDevice, stream);
  k_cvt16<<<(NT * CSD) / 256, 256, 0, stream>>>(sc16, single_cond);

  for (int l = 0; l < 8; ++l) {
    const int sb = l / 4, s = l % 4;
    if (s == 0) {
      k_wtr<<<dim3(CPD / 64, SBH / 64), 256, 0, stream>>>(
          pair_w + (size_t)sb * CPD * SBH, SBH, CPD, pwT16);
      k_pl<<<(NT * NT) / 64, 256, 0, stream>>>(pair_cond, pair_ln, pwT16, plT16);
    }

    // ---- attention adaLN ----
    k_ln<<<NT, 256, 0, stream>>>(single_cond, CSD, attn_cln + (size_t)l * CSD, nullptr, cn16);
    k_ln<<<NT, 256, 0, stream>>>(a, CD, nullptr, lnA, nullptr);
    gemm(G, 1, cn16, CSD, attn_cs_w + (size_t)l * CSD * CD, CD,
         attn_cs_b + (size_t)l * CD, s1, nullptr, CD, nullptr, NT, CD, CSD);
    gemm(G, 0, cn16, CSD, attn_cb_w + (size_t)l * CSD * CD, CD,
         nullptr, b1, nullptr, CD, nullptr, NT, CD, CSD);
    k_make_x<<<(NT * CD) / 256, 256, 0, stream>>>(x16, s1, lnA, b1);

    // ---- qkv + gate ----
    gemm(G, 2, x16, CD, q_w + (size_t)l * CD * CD, CD, q_b + (size_t)l * CD,
         nullptr, q16, CD, nullptr, NT, CD, CD);
    gemm(G, 2, x16, CD, k_w + (size_t)l * CD * CD, CD, nullptr,
         nullptr, k16, CD, nullptr, NT, CD, CD);
    gemm(G, 4, x16, CD, v_w + (size_t)l * CD * CD, CD, nullptr,
         nullptr, vT16, NT, nullptr, NT, CD, CD);
    gemm(G, 1, x16, CD, g_w + (size_t)l * CD * CD, CD, g_b + (size_t)l * CD,
         gate, nullptr, CD, nullptr, NT, CD, CD);

    // ---- attention core ----
    k_scores<<<dim3(NT / 64, NT / 64, NH), 256, 0, stream>>>(q16, k16, mask, plT16, s, logit);
    k_softmax<<<NH * NT, 256, 0, stream>>>(logit, wts16);
    k_wa<<<dim3(NT / 64, NH), 256, 0, stream>>>(wts16, vT16, gate, wa16);

    // ---- output proj + adaptive zero + residual ----
    gemm(G, 1, sc16, CSD, az_w + (size_t)l * CSD * CD, CD, az_b + (size_t)l * CD,
         az, nullptr, CD, nullptr, NT, CD, CSD);
    gemm(G, 3, wa16, CD, o_w + (size_t)l * CD * CD, CD, nullptr,
         a, nullptr, CD, az, NT, CD, CD);

    // ---- FFN adaLN ----
    k_ln<<<NT, 256, 0, stream>>>(single_cond, CSD, f_cln + (size_t)l * CSD, nullptr, cn16);
    k_ln<<<NT, 256, 0, stream>>>(a, CD, nullptr, lnA, nullptr);
    gemm(G, 1, cn16, CSD, f_cs_w + (size_t)l * CSD * CD, CD,
         f_cs_b + (size_t)l * CD, s1, nullptr, CD, nullptr, NT, CD, CSD);
    gemm(G, 0, cn16, CSD, f_cb_w + (size_t)l * CSD * CD, CD,
         nullptr, b1, nullptr, CD, nullptr, NT, CD, CSD);
    k_make_x<<<(NT * CD) / 256, 256, 0, stream>>>(x16, s1, lnA, b1);

    // ---- SwiGLU FFN + adaptive zero + residual ----
    gemm(G, 0, x16, CD, t1_w + (size_t)l * CD * 2 * FFD, 2 * FFD, nullptr,
         u, nullptr, 2 * FFD, nullptr, NT, 2 * FFD, CD);
    k_glu<<<(NT * FFD) / 256, 256, 0, stream>>>(c16, u);
    gemm(G, 1, sc16, CSD, f_az_w + (size_t)l * CSD * CD, CD, f_az_b + (size_t)l * CD,
         az, nullptr, CD, nullptr, NT, CD, CSD);
    gemm(G, 3, c16, FFD, t2_w + (size_t)l * FFD * CD, CD, nullptr,
         a, nullptr, CD, az, NT, CD, FFD);
  }
}
